// MultiHeadSTEVESA_54554674594024
// MI455X (gfx1250) — matrix operand order repaired and checked by reading, not compiled, not run on hardware
//
#include <hip/hip_runtime.h>

typedef __attribute__((ext_vector_type(16))) __bf16 v16bf;
typedef __attribute__((ext_vector_type(8)))  __bf16 v8bf;
typedef __attribute__((ext_vector_type(8)))  float  v8f;

#define LN_EPS 1e-5f
#define EPSC   1e-8f

// ---------------- WMMA fragment helpers (CDNA5 16x16x32 bf16 layouts) ----------------

__device__ __forceinline__ v16bf load_a_frag(const __bf16* base, int stride, int mrow, int k0, int lane) {
  // A (MxK, 16x32 bf16): lanes 0-15 -> K = k0+0..7 (V0-3), k0+16..23 (V4-7)
  //                      lanes16-31 -> K = k0+8..15,        k0+24..31
  int mm = mrow + (lane & 15);
  int kh = (lane < 16) ? 0 : 8;
  const __bf16* p = base + (size_t)mm * stride + k0 + kh;
  union { v16bf v; v8bf h[2]; } u;
  u.h[0] = *(const v8bf*)(p);
  u.h[1] = *(const v8bf*)(p + 16);
  return u.v;
}

__device__ __forceinline__ v16bf load_b_frag(const __bf16* bt, int stride, int n0, int k0, int lane) {
  // B (KxN) supplied as B^T row-major: Bt[n][k]. lanes 0-15: K=k0+0..7, k0+16..23; lanes16-31: K=k0+8..15, k0+24..31
  int n = n0 + (lane & 15);
  int kb = k0 + ((lane < 16) ? 0 : 8);
  const __bf16* p = bt + (size_t)n * stride + kb;
  union { v16bf v; v8bf h[2]; } u;
  u.h[0] = *(const v8bf*)(p);
  u.h[1] = *(const v8bf*)(p + 16);
  return u.v;
}

__device__ __forceinline__ v8f wmma_bf16(v16bf a, v16bf b, v8f c) {
  return __builtin_amdgcn_wmma_f32_16x16x32_bf16(false, a, false, b, (short)0, c, false, false);
}

__device__ __forceinline__ v8f v8f_zero() {
  v8f z = {0.f,0.f,0.f,0.f,0.f,0.f,0.f,0.f};
  return z;
}

// One A row-tile (mt) against NT consecutive B column tiles: A fragment loaded
// once per k-step and reused for NT WMMAs.
template <int NT>
__device__ __forceinline__ void gemm_tiles(const __bf16* A, int as, const __bf16* Bt, int bs,
                                           int K, int mt, int nt0, int lane, v8f* acc) {
  for (int k0 = 0; k0 < K; k0 += 32) {
    v16bf af = load_a_frag(A, as, mt * 16, k0, lane);
#pragma unroll
    for (int j = 0; j < NT; j++) {
      v16bf bf = load_b_frag(Bt, bs, (nt0 + j) * 16, k0, lane);
      acc[j] = wmma_bf16(af, bf, acc[j]);
    }
  }
}

// ---------------- prep kernels ----------------

// dst[n*K + k] = bf16(src[k*N + n])   (B^T image for WMMA B operand)
__global__ void k_transpose_bf16(const float* __restrict__ src, __bf16* __restrict__ dst, int K, int N) {
  int idx = blockIdx.x * 256 + threadIdx.x;
  if (idx >= K * N) return;
  int n = idx / K, k = idx - n * K;
  dst[idx] = (__bf16)src[(size_t)k * N + n];
}

// slots ws [B][32][256] f32 (rows 24..31 zero padding)
__global__ void k_slots_init(const float* __restrict__ mu, float* __restrict__ slots) {
  int idx = blockIdx.x * 256 + threadIdx.x;  // B*32*256
  int c = idx & 255, r = (idx >> 8) & 31;
  slots[idx] = (r < 24) ? mu[r * 256 + c] : 0.f;
}

// vsum[b*256 + h*64 + d] = sum_n vT[(b,h,d)][n]
__global__ void k_vsum(const __bf16* __restrict__ vT, float* __restrict__ vsum) {
  int bid = blockIdx.x;  // (b*4+h)*64+d
  const __bf16* row = vT + (size_t)bid * 4096;
  float s = 0.f;
  for (int n = threadIdx.x; n < 4096; n += 256) s += (float)row[n];
  __shared__ float red[256];
  red[threadIdx.x] = s; __syncthreads();
  for (int off = 128; off > 0; off >>= 1) {
    if (threadIdx.x < off) red[threadIdx.x] += red[threadIdx.x + off];
    __syncthreads();
  }
  if (threadIdx.x == 0) vsum[bid] = red[0];
}

// ---------------- token pipeline: pos + LN + MLP(2x) + LN + k,v ----------------

__global__ __launch_bounds__(256) void k_tokens(
    const float* __restrict__ inp, const float* __restrict__ pos_w, const float* __restrict__ pos_b,
    const float* __restrict__ g_in, const float* __restrict__ b_in,
    const __bf16* __restrict__ W1T, const float* __restrict__ b1,
    const __bf16* __restrict__ W2T, const float* __restrict__ b2,
    const float* __restrict__ g_inp, const float* __restrict__ b_inp,
    const __bf16* __restrict__ WkT, const __bf16* __restrict__ WvT,
    __bf16* __restrict__ kOut, __bf16* __restrict__ vTOut) {
  __shared__ __bf16 xa[32][264];
  __shared__ __bf16 xb[32][264];      // also reused as k-tile staging
  __shared__ __bf16 vt[256][40];      // v tile transposed [col][token]
  __shared__ float redS[32][8], redQ[32][8];
  __shared__ float mu_[32], rs_[32];

  int tid = threadIdx.x, lane = tid & 31, wv = tid >> 5;
  int blk = blockIdx.x;
  int b = blk >> 8;             // 256 tiles of 32 tokens per batch
  int n0 = (blk & 255) * 32;

  // stage 1: load + positional embed + LN_in partials (32 features/thread)
  int tok = tid & 31, grp = tid >> 5;
  int n = n0 + tok;
  int pr = n >> 6, pc = n & 63;
  float gx = (pc + 0.5f) * (1.f / 64.f), gy = (pr + 0.5f) * (1.f / 64.f);
  float vbuf[32];
  {
    float s = 0.f, s2 = 0.f;
    for (int i = 0; i < 32; i++) {
      int c = grp * 32 + i;
      float val = inp[((size_t)(b * 256 + c)) * 4096 + n]
                + pos_w[c] * gx + pos_w[256 + c] * gy
                + pos_w[512 + c] * (1.f - gx) + pos_w[768 + c] * (1.f - gy) + pos_b[c];
      vbuf[i] = val; s += val; s2 += val * val;
    }
    redS[tok][grp] = s; redQ[tok][grp] = s2;
  }
  __syncthreads();
  if (tid < 32) {
    float a = 0.f, q = 0.f;
    for (int i = 0; i < 8; i++) { a += redS[tid][i]; q += redQ[tid][i]; }
    float m = a * (1.f / 256.f), v = q * (1.f / 256.f) - m * m;
    mu_[tid] = m; rs_[tid] = rsqrtf(v + LN_EPS);
  }
  __syncthreads();
  {
    float m = mu_[tok], r = rs_[tok];
    for (int i = 0; i < 32; i++) {
      int c = grp * 32 + i;
      xa[tok][c] = (__bf16)(((vbuf[i] - m) * r) * g_in[c] + b_in[c]);
    }
  }
  __syncthreads();

  int mt = wv & 1, nt0 = (wv >> 1) * 4;   // wave owns one row-tile, 4 col-tiles

  // GEMM1: relu(xa @ W1 + b1) -> xb
  {
    v8f acc[4];
#pragma unroll
    for (int j = 0; j < 4; j++) acc[j] = v8f_zero();
    gemm_tiles<4>(&xa[0][0], 264, W1T, 256, 256, mt, nt0, lane, acc);
    int rbase = mt * 16 + ((lane < 16) ? 0 : 8);
#pragma unroll
    for (int j = 0; j < 4; j++) {
      int col = (nt0 + j) * 16 + (lane & 15);
      float bias = b1[col];
#pragma unroll
      for (int vr = 0; vr < 8; vr++) {
        float x = acc[j][vr] + bias;
        xb[rbase + vr][col] = (__bf16)(x > 0.f ? x : 0.f);
      }
    }
  }
  if (tid < 32) { redS[tid][0] = 0.f; redQ[tid][0] = 0.f; }
  __syncthreads();

  // GEMM2: x2 = xb @ W2 + b2, then LN_inp -> xa
  {
    v8f a2[4];
#pragma unroll
    for (int j = 0; j < 4; j++) a2[j] = v8f_zero();
    gemm_tiles<4>(&xb[0][0], 264, W2T, 256, 256, mt, nt0, lane, a2);
    int rbase = mt * 16 + ((lane < 16) ? 0 : 8);
#pragma unroll
    for (int j = 0; j < 4; j++) {
      float bias = b2[(nt0 + j) * 16 + (lane & 15)];
#pragma unroll
      for (int vr = 0; vr < 8; vr++) {
        float x = a2[j][vr] + bias; a2[j][vr] = x;
        float sx = x, sq = x * x;
        for (int msk = 1; msk < 16; msk <<= 1) {
          sx += __shfl_xor(sx, msk, 32);
          sq += __shfl_xor(sq, msk, 32);
        }
        if ((lane & 15) == 0) {
          atomicAdd(&redS[rbase + vr][0], sx);
          atomicAdd(&redQ[rbase + vr][0], sq);
        }
      }
    }
    __syncthreads();
    if (tid < 32) {
      float a = redS[tid][0], q = redQ[tid][0];
      float m = a * (1.f / 256.f), v = q * (1.f / 256.f) - m * m;
      mu_[tid] = m; rs_[tid] = rsqrtf(v + LN_EPS);
    }
    __syncthreads();
#pragma unroll
    for (int j = 0; j < 4; j++) {
      int col = (nt0 + j) * 16 + (lane & 15);
      float g = g_inp[col], bb = b_inp[col];
#pragma unroll
      for (int vr = 0; vr < 8; vr++) {
        int row = rbase + vr;
        xa[row][col] = (__bf16)(((a2[j][vr] - mu_[row]) * rs_[row]) * g + bb);
      }
    }
  }
  __syncthreads();

  // GEMM k (scaled 1/16) and v: shared A fragment feeds both weight streams
  {
    v8f ak[4], av[4];
#pragma unroll
    for (int j = 0; j < 4; j++) { ak[j] = v8f_zero(); av[j] = v8f_zero(); }
    for (int k0 = 0; k0 < 256; k0 += 32) {
      v16bf af = load_a_frag(&xa[0][0], 264, mt * 16, k0, lane);
#pragma unroll
      for (int j = 0; j < 4; j++) {
        v16bf bfk = load_b_frag(WkT, 256, (nt0 + j) * 16, k0, lane);
        ak[j] = wmma_bf16(af, bfk, ak[j]);
        v16bf bfv = load_b_frag(WvT, 256, (nt0 + j) * 16, k0, lane);
        av[j] = wmma_bf16(af, bfv, av[j]);
      }
    }
    int rbase = mt * 16 + ((lane < 16) ? 0 : 8);
#pragma unroll
    for (int j = 0; j < 4; j++) {
      int col = (nt0 + j) * 16 + (lane & 15);
#pragma unroll
      for (int vr = 0; vr < 8; vr++) {
        xb[rbase + vr][col] = (__bf16)(ak[j][vr] * 0.0625f);  // k staging (row-major)
        vt[col][rbase + vr] = (__bf16)av[j][vr];              // v staging (transposed)
      }
    }
  }
  __syncthreads();
  // chunked 64B stores: k[b][h][n][d] rows and vT[b][h][d][n] rows
  {
    int token = tid & 31, h = (tid >> 5) & 3, half = tid >> 7;
    const __bf16* src = &xb[token][h * 64 + half * 32];
    __bf16* dst = kOut + (((size_t)(b * 4 + h) * 4096) + n0 + token) * 64 + half * 32;
#pragma unroll
    for (int q = 0; q < 4; q++) ((v8bf*)dst)[q] = ((const v8bf*)src)[q];
  }
  {
    int col = tid, h = col >> 6, d = col & 63;
    const __bf16* src = &vt[col][0];
    __bf16* dst = vTOut + (((size_t)(b * 4 + h) * 64) + d) * 4096 + n0;
#pragma unroll
    for (int q = 0; q < 4; q++) ((v8bf*)dst)[q] = ((const v8bf*)src)[q];
  }
}

// ---------------- attention iteration: q, logits, joint softmax, upd ----------------

__global__ __launch_bounds__(256) void k_attn(
    const float* __restrict__ slots, const float* __restrict__ g_s, const float* __restrict__ b_s,
    const __bf16* __restrict__ WqT, const __bf16* __restrict__ kAll, const __bf16* __restrict__ vTAll,
    const float* __restrict__ vsum, __bf16* __restrict__ updOut) {
  __shared__ __bf16 sa[32][264];
  __shared__ __bf16 qb[4][32][72];
  __shared__ float  L[32][104];
  __shared__ __bf16 pT[4][32][40];
  __shared__ float  P[4][32];
  __shared__ float  VS[4][64];
  __shared__ float  redS[32][8], redQ[32][8], mu_[32], rs_[32];

  int tid = threadIdx.x, lane = tid & 31, wv = tid >> 5;
  int b = blockIdx.x;
  const float* sl = slots + (size_t)b * 32 * 256;

  // LN_slot(slots) -> sa
  {
    int row = tid & 31, grp = tid >> 5;
    float s = 0.f, s2 = 0.f; float buf[32];
    for (int i = 0; i < 32; i++) {
      float x = sl[row * 256 + grp * 32 + i];
      buf[i] = x; s += x; s2 += x * x;
    }
    redS[row][grp] = s; redQ[row][grp] = s2;
    __syncthreads();
    if (tid < 32) {
      float a = 0.f, q = 0.f;
      for (int i = 0; i < 8; i++) { a += redS[tid][i]; q += redQ[tid][i]; }
      float m = a * (1.f / 256.f), v = q * (1.f / 256.f) - m * m;
      mu_[tid] = m; rs_[tid] = rsqrtf(v + LN_EPS);
    }
    __syncthreads();
    float m = mu_[row], r = rs_[row];
    for (int i = 0; i < 32; i++) {
      int c = grp * 32 + i;
      sa[row][c] = (__bf16)(((buf[i] - m) * r) * g_s[c] + b_s[c]);
    }
  }
  __syncthreads();

  // q = sa @ Wq -> qb[h][s][d]
  {
    int mt = wv & 1, nt0 = (wv >> 1) * 4;
    v8f acc[4];
#pragma unroll
    for (int j = 0; j < 4; j++) acc[j] = v8f_zero();
    gemm_tiles<4>(&sa[0][0], 264, WqT, 256, 256, mt, nt0, lane, acc);
    int rbase = mt * 16 + ((lane < 16) ? 0 : 8);
#pragma unroll
    for (int j = 0; j < 4; j++) {
      int col = (nt0 + j) * 16 + (lane & 15);
      int h = col >> 6, d = col & 63;
#pragma unroll
      for (int vr = 0; vr < 8; vr++) qb[h][rbase + vr][d] = (__bf16)acc[j][vr];
    }
  }
  if (tid < 128) P[tid >> 5][tid & 31] = 0.f;
  for (int i = tid; i < 4 * 32 * 40; i += 256) (&pT[0][0][0])[i] = (__bf16)0.f;
  (&VS[0][0])[tid] = vsum[b * 256 + tid];
  __syncthreads();

  int h_w = wv & 3, dh = wv >> 2;
  const __bf16* vB = vTAll + (size_t)(b * 4 + h_w) * 64 * 4096;
  const __bf16* kB = kAll + (size_t)(b * 4 + h_w) * 4096 * 64;
  v8f U[2][2];
  U[0][0] = v8f_zero(); U[0][1] = v8f_zero(); U[1][0] = v8f_zero(); U[1][1] = v8f_zero();

  for (int nc = 0; nc < 4096; nc += 32) {
    // prefetch next chunk of k and vT (CDNA5 global_prefetch path)
    if (nc + 32 < 4096) {
      __builtin_prefetch(kB + (size_t)(nc + 32) * 64 + lane * 64, 0, 3);
      __builtin_prefetch(vB + (size_t)(dh * 32 + lane) * 4096 + nc + 32, 0, 3);
    }
    // logits: wave -> (h = wv&3, st = wv>>2); B fragment hoisted across row-tiles
    {
      int st = dh;
      const __bf16* kT0 = kB + (size_t)nc * 64;
      v8f acc[2]; acc[0] = v8f_zero(); acc[1] = v8f_zero();
      for (int k0 = 0; k0 < 64; k0 += 32) {
        v16bf bf = load_b_frag(&qb[h_w][0][0], 72, st * 16, k0, lane);
#pragma unroll
        for (int mt = 0; mt < 2; mt++) {
          v16bf af = load_a_frag(kT0, 64, mt * 16, k0, lane);
          acc[mt] = wmma_bf16(af, bf, acc[mt]);
        }
      }
      int scol = st * 16 + (lane & 15);
      if (scol < 24) {
#pragma unroll
        for (int mt = 0; mt < 2; mt++) {
          int rb = mt * 16 + ((lane < 16) ? 0 : 8);
#pragma unroll
          for (int vr = 0; vr < 8; vr++) L[rb + vr][h_w * 24 + scol] = acc[mt][vr];
        }
      }
    }
    __syncthreads();
    // joint softmax over (h,s)=96 per token: 8 threads per token, shuffle reductions
    {
      int token = tid >> 3, sub = tid & 7;
      float lx[12];
      float mx = -1e30f;
#pragma unroll
      for (int i = 0; i < 12; i++) { lx[i] = L[token][sub * 12 + i]; mx = fmaxf(mx, lx[i]); }
      for (int m = 1; m < 8; m <<= 1) mx = fmaxf(mx, __shfl_xor(mx, m, 32));
      float ss = 0.f;
#pragma unroll
      for (int i = 0; i < 12; i++) { float e = __expf(lx[i] - mx); lx[i] = e; ss += e; }
      for (int m = 1; m < 8; m <<= 1) ss += __shfl_xor(ss, m, 32);
      float inv = 1.f / ss;
#pragma unroll
      for (int i = 0; i < 12; i++) {
        int j = sub * 12 + i;
        int hh = j / 24, sidx = j - hh * 24;
        pT[hh][sidx][token] = (__bf16)(lx[i] * inv);
      }
    }
    __syncthreads();
    // P accumulation
    if (tid < 96) {
      int h = tid / 24, sidx = tid - h * 24;
      float acc = 0.f;
      for (int nl = 0; nl < 32; nl++) acc += (float)pT[h][sidx][nl];
      P[h][sidx] += acc;
    }
    // upd accumulation: U += p^T @ v   (A = pT[h], K = 32 tokens)
#pragma unroll
    for (int mt = 0; mt < 2; mt++) {
      v16bf af = load_a_frag(&pT[h_w][0][0], 40, mt * 16, 0, lane);
#pragma unroll
      for (int j = 0; j < 2; j++) {
        int dt = dh * 2 + j;
        v16bf bf = load_b_frag(vB, 4096, dt * 16, nc, lane);
        U[mt][j] = wmma_bf16(af, bf, U[mt][j]);
      }
    }
    __syncthreads();
  }

  // finalize: upd = (U + EPS*Vsum) / (P + N*EPS)
  __bf16* updB = updOut + (size_t)b * 32 * 256;
#pragma unroll
  for (int mt = 0; mt < 2; mt++) {
#pragma unroll
    for (int j = 0; j < 2; j++) {
      int dt = dh * 2 + j;
      int d = dt * 16 + (lane & 15);
      int rbase = mt * 16 + ((lane < 16) ? 0 : 8);
      float vs = VS[h_w][d];
#pragma unroll
      for (int vr = 0; vr < 8; vr++) {
        int sidx = rbase + vr;
        float u = (U[mt][j][vr] + EPSC * vs) / (P[h_w][sidx] + 4096.f * EPSC);
        updB[sidx * 256 + h_w * 64 + d] = (__bf16)u;
      }
    }
  }
}

// ---------------- GRU + slot MLP ----------------

__global__ __launch_bounds__(256) void k_gru(
    float* __restrict__ slots, const __bf16* __restrict__ upd,
    const __bf16* __restrict__ WiT, const __bf16* __restrict__ WhT,
    const float* __restrict__ gbi, const float* __restrict__ gbh,
    const float* __restrict__ gm, const float* __restrict__ bm,
    const __bf16* __restrict__ M1T, const float* __restrict__ m1b,
    const __bf16* __restrict__ M2T, const float* __restrict__ m2b,
    __bf16* __restrict__ hid) {
  __shared__ __bf16 hpA[32][264];     // raw h_prev (bf16), later reused as LN'd A tile
  __shared__ __bf16 R[32][264];       // r gate
  __shared__ __bf16 sn[32][264];      // new slots (pre-MLP), bf16
  __shared__ float  redS[32][8], redQ[32][8], mu_[32], rs_[32];

  int tid = threadIdx.x, lane = tid & 31, wv = tid >> 5;
  int b = blockIdx.x;
  float* slB = slots + (size_t)b * 32 * 256;
  const __bf16* updB = upd + (size_t)b * 32 * 256;
  __bf16* hidB = hid + (size_t)b * 32 * 1024;

  int row = tid & 31, grp = tid >> 5;
  for (int i = 0; i < 32; i++) {
    int c = grp * 32 + i;
    hpA[row][c] = (__bf16)slB[row * 256 + c];
  }
  __syncthreads();

  // phase 1: r = sigmoid(xr + hr); A fragments shared between the two gate GEMMs
  for (int i = 0; i < 4; i++) {
    int jb = wv * 4 + i; int mt = jb & 1, ct = jb >> 1;
    v8f ax = v8f_zero(), ah = v8f_zero();
    for (int k0 = 0; k0 < 256; k0 += 32) {
      v16bf aU = load_a_frag(updB, 256, mt * 16, k0, lane);
      v16bf aH = load_a_frag(&hpA[0][0], 264, mt * 16, k0, lane);
      v16bf bI = load_b_frag(WiT, 256, ct * 16, k0, lane);
      ax = wmma_bf16(aU, bI, ax);
      v16bf bH = load_b_frag(WhT, 256, ct * 16, k0, lane);
      ah = wmma_bf16(aH, bH, ah);
    }
    int j = ct * 16 + (lane & 15);
    float bi = gbi[j], bh = gbh[j];
    int rbase = mt * 16 + ((lane < 16) ? 0 : 8);
#pragma unroll
    for (int vr = 0; vr < 8; vr++) {
      float r = 1.f / (1.f + __expf(-(ax[vr] + bi + ah[vr] + bh)));
      R[rbase + vr][j] = (__bf16)r;
    }
  }
  __syncthreads();

  // phase 2: z, n, slot update -> sn (4 WMMAs per shared A-fragment pair)
  for (int i = 0; i < 4; i++) {
    int jb = wv * 4 + i; int mt = jb & 1, ct = jb >> 1;
    v8f xz = v8f_zero(), hz = v8f_zero(), xn = v8f_zero(), hn = v8f_zero();
    for (int k0 = 0; k0 < 256; k0 += 32) {
      v16bf aU = load_a_frag(updB, 256, mt * 16, k0, lane);
      v16bf aH = load_a_frag(&hpA[0][0], 264, mt * 16, k0, lane);
      v16bf bIz = load_b_frag(WiT, 256, (16 + ct) * 16, k0, lane);
      xz = wmma_bf16(aU, bIz, xz);
      v16bf bHz = load_b_frag(WhT, 256, (16 + ct) * 16, k0, lane);
      hz = wmma_bf16(aH, bHz, hz);
      v16bf bIn = load_b_frag(WiT, 256, (32 + ct) * 16, k0, lane);
      xn = wmma_bf16(aU, bIn, xn);
      v16bf bHn = load_b_frag(WhT, 256, (32 + ct) * 16, k0, lane);
      hn = wmma_bf16(aH, bHn, hn);
    }
    int j = ct * 16 + (lane & 15);
    float biz = gbi[256 + j], bhz = gbh[256 + j];
    float bin = gbi[512 + j], bhn = gbh[512 + j];
    int rbase = mt * 16 + ((lane < 16) ? 0 : 8);
#pragma unroll
    for (int vr = 0; vr < 8; vr++) {
      int rw = rbase + vr;
      float z = 1.f / (1.f + __expf(-(xz[vr] + biz + hz[vr] + bhz)));
      float nv = tanhf(xn[vr] + bin + (float)R[rw][j] * (hn[vr] + bhn));
      float hp = (float)hpA[rw][j];
      sn[rw][j] = (__bf16)((1.f - z) * nv + z * hp);
    }
  }
  __syncthreads();

  // phase 3: LN_mlp(sn) -> hpA (reused as A tile)
  {
    float s = 0.f, s2 = 0.f; float buf[32];
    for (int i = 0; i < 32; i++) {
      float x = (float)sn[row][grp * 32 + i];
      buf[i] = x; s += x; s2 += x * x;
    }
    redS[row][grp] = s; redQ[row][grp] = s2;
    __syncthreads();
    if (tid < 32) {
      float a = 0.f, q = 0.f;
      for (int i = 0; i < 8; i++) { a += redS[tid][i]; q += redQ[tid][i]; }
      float m = a * (1.f / 256.f), v = q * (1.f / 256.f) - m * m;
      mu_[tid] = m; rs_[tid] = rsqrtf(v + LN_EPS);
    }
    __syncthreads();
    float m = mu_[row], r = rs_[row];
    for (int i = 0; i < 32; i++) {
      int c = grp * 32 + i;
      hpA[row][c] = (__bf16)(((buf[i] - m) * r) * gm[c] + bm[c]);
    }
  }
  __syncthreads();

  // phase 3b: hidden = relu(lnA @ mlp_w1 + b1) -> workspace (bf16 [32][1024])
  for (int g = 0; g < 2; g++) {
    int mt = wv & 1, nt0 = (wv >> 1) * 16 + g * 8;
    v8f acc[8];
#pragma unroll
    for (int j = 0; j < 8; j++) acc[j] = v8f_zero();
    gemm_tiles<8>(&hpA[0][0], 264, M1T, 256, 256, mt, nt0, lane, acc);
    int rbase = mt * 16 + ((lane < 16) ? 0 : 8);
#pragma unroll
    for (int j = 0; j < 8; j++) {
      int col = (nt0 + j) * 16 + (lane & 15);
      float bias = m1b[col];
#pragma unroll
      for (int vr = 0; vr < 8; vr++) {
        float x = acc[j][vr] + bias;
        hidB[(size_t)(rbase + vr) * 1024 + col] = (__bf16)(x > 0.f ? x : 0.f);
      }
    }
  }
  __threadfence();
  __syncthreads();

  // phase 4: slots = sn + hidden @ mlp_w2 + b2
  {
    int mt = wv & 1, nt0 = (wv >> 1) * 4;
    v8f acc[4];
#pragma unroll
    for (int j = 0; j < 4; j++) acc[j] = v8f_zero();
    gemm_tiles<4>(hidB, 1024, M2T, 1024, 1024, mt, nt0, lane, acc);
    int rbase = mt * 16 + ((lane < 16) ? 0 : 8);
#pragma unroll
    for (int j = 0; j < 4; j++) {
      int col = (nt0 + j) * 16 + (lane & 15);
      float bias = m2b[col];
#pragma unroll
      for (int vr = 0; vr < 8; vr++) {
        int rw = rbase + vr;
        slB[rw * 256 + col] = acc[j][vr] + bias + (float)sn[rw][col];
      }
    }
  }
}

// ---------------- output projection ----------------

__global__ __launch_bounds__(256) void k_out(
    const float* __restrict__ slots, const float* __restrict__ g, const float* __restrict__ bb,
    const __bf16* __restrict__ OwT, const float* __restrict__ ob, float* __restrict__ out) {
  __shared__ __bf16 lnA[32][264];
  __shared__ float redS[32][8], redQ[32][8], mu_[32], rs_[32];
  int tid = threadIdx.x, lane = tid & 31, wv = tid >> 5;
  int b = blockIdx.x;
  const float* sl = slots + (size_t)b * 32 * 256;
  int row = tid & 31, grp = tid >> 5;
  {
    float s = 0.f, s2 = 0.f; float buf[32];
    for (int i = 0; i < 32; i++) {
      float x = sl[row * 256 + grp * 32 + i];
      buf[i] = x; s += x; s2 += x * x;
    }
    redS[row][grp] = s; redQ[row][grp] = s2;
    __syncthreads();
    if (tid < 32) {
      float a = 0.f, q = 0.f;
      for (int i = 0; i < 8; i++) { a += redS[tid][i]; q += redQ[tid][i]; }
      float m = a * (1.f / 256.f), v = q * (1.f / 256.f) - m * m;
      mu_[tid] = m; rs_[tid] = rsqrtf(v + LN_EPS);
    }
    __syncthreads();
    float m = mu_[row], r = rs_[row];
    for (int i = 0; i < 32; i++) {
      int c = grp * 32 + i;
      lnA[row][c] = (__bf16)(((buf[i] - m) * r) * g[c] + bb[c]);
    }
  }
  __syncthreads();
  {
    int mt = wv & 1, nt0 = (wv >> 1) * 4;
    v8f acc[4];
#pragma unroll
    for (int j = 0; j < 4; j++) acc[j] = v8f_zero();
    gemm_tiles<4>(&lnA[0][0], 264, OwT, 256, 256, mt, nt0, lane, acc);
    int rbase = mt * 16 + ((lane < 16) ? 0 : 8);
#pragma unroll
    for (int j = 0; j < 4; j++) {
      int col = (nt0 + j) * 16 + (lane & 15);
      float bias = ob[col];
#pragma unroll
      for (int vr = 0; vr < 8; vr++) {
        int s = rbase + vr;
        if (s < 24) out[((size_t)b * 24 + s) * 256 + col] = acc[j][vr] + bias;
      }
    }
  }
}

// ---------------- host launcher ----------------

extern "C" void kernel_launch(void* const* d_in, const int* in_sizes, int n_in,
                              void* d_out, int out_size, void* d_ws, size_t ws_size,
                              hipStream_t stream) {
  (void)in_sizes; (void)n_in; (void)out_size; (void)ws_size;
  const float* inputs   = (const float*)d_in[0];
  const float* pos_w    = (const float*)d_in[1];
  const float* pos_b    = (const float*)d_in[2];
  const float* ln_in_g  = (const float*)d_in[3];
  const float* ln_in_b  = (const float*)d_in[4];
  const float* mlp_in_w1= (const float*)d_in[5];
  const float* mlp_in_b1= (const float*)d_in[6];
  const float* mlp_in_w2= (const float*)d_in[7];
  const float* mlp_in_b2= (const float*)d_in[8];
  const float* slot_mu  = (const float*)d_in[9];
  const float* ln_inp_g = (const float*)d_in[10];
  const float* ln_inp_b = (const float*)d_in[11];
  const float* ln_slot_g= (const float*)d_in[12];
  const float* ln_slot_b= (const float*)d_in[13];
  const float* ln_mlp_g = (const float*)d_in[14];
  const float* ln_mlp_b = (const float*)d_in[15];
  const float* Wq       = (const float*)d_in[16];
  const float* Wk       = (const float*)d_in[17];
  const float* Wv       = (const float*)d_in[18];
  const float* gru_wi   = (const float*)d_in[19];
  const float* gru_wh   = (const float*)d_in[20];
  const float* gru_bi   = (const float*)d_in[21];
  const float* gru_bh   = (const float*)d_in[22];
  const float* mlp_w1   = (const float*)d_in[23];
  const float* mlp_b1   = (const float*)d_in[24];
  const float* mlp_w2   = (const float*)d_in[25];
  const float* mlp_b2   = (const float*)d_in[26];
  const float* ln_out_g = (const float*)d_in[27];
  const float* ln_out_b = (const float*)d_in[28];
  const float* out_w    = (const float*)d_in[29];
  const float* out_b    = (const float*)d_in[30];
  float* out = (float*)d_out;

  char* ws = (char*)d_ws;
  size_t off = 0;
  auto alloc = [&](size_t bytes) -> void* {
    void* p = ws + off;
    off = (off + bytes + 255) & ~(size_t)255;
    return p;
  };
  __bf16* W1T = (__bf16*)alloc((size_t)256 * 256 * 2);
  __bf16* W2T = (__bf16*)alloc((size_t)256 * 256 * 2);
  __bf16* WkT = (__bf16*)alloc((size_t)256 * 256 * 2);
  __bf16* WvT = (__bf16*)alloc((size_t)256 * 256 * 2);
  __bf16* WqT = (__bf16*)alloc((size_t)256 * 256 * 2);
  __bf16* WiT = (__bf16*)alloc((size_t)768 * 256 * 2);
  __bf16* WhT = (__bf16*)alloc((size_t)768 * 256 * 2);
  __bf16* M1T = (__bf16*)alloc((size_t)1024 * 256 * 2);
  __bf16* M2T = (__bf16*)alloc((size_t)256 * 1024 * 2);
  __bf16* OwT = (__bf16*)alloc((size_t)256 * 256 * 2);
  float*  slotsWs = (float*)alloc((size_t)64 * 32 * 256 * 4);
  __bf16* updWs   = (__bf16*)alloc((size_t)64 * 32 * 256 * 2);
  __bf16* hidWs   = (__bf16*)alloc((size_t)64 * 32 * 1024 * 2);
  float*  vsumWs  = (float*)alloc((size_t)64 * 256 * 4);
  __bf16* kWs     = (__bf16*)alloc((size_t)64 * 4 * 4096 * 64 * 2);
  __bf16* vTWs    = (__bf16*)alloc((size_t)64 * 4 * 4096 * 64 * 2);

  auto T = [&](const float* src, __bf16* dst, int K, int N) {
    int total = K * N;
    k_transpose_bf16<<<(total + 255) / 256, 256, 0, stream>>>(src, dst, K, N);
  };
  T(mlp_in_w1, W1T, 256, 256);
  T(mlp_in_w2, W2T, 256, 256);
  T(Wk, WkT, 256, 256);
  T(Wv, WvT, 256, 256);
  T(Wq, WqT, 256, 256);
  T(gru_wi, WiT, 256, 768);
  T(gru_wh, WhT, 256, 768);
  T(mlp_w1, M1T, 256, 1024);
  T(mlp_w2, M2T, 1024, 256);
  T(out_w, OwT, 256, 256);

  k_slots_init<<<(64 * 32 * 256) / 256, 256, 0, stream>>>(slot_mu, slotsWs);

  k_tokens<<<64 * 128, 256, 0, stream>>>(inputs, pos_w, pos_b, ln_in_g, ln_in_b,
                                         W1T, mlp_in_b1, W2T, mlp_in_b2,
                                         ln_inp_g, ln_inp_b, WkT, WvT, kWs, vTWs);

  k_vsum<<<64 * 4 * 64, 256, 0, stream>>>(vTWs, vsumWs);

  for (int it = 0; it < 3; it++) {
    k_attn<<<64, 256, 0, stream>>>(slotsWs, ln_slot_g, ln_slot_b, WqT, kWs, vTWs,
                                   vsumWs, updWs);
    k_gru<<<64, 256, 0, stream>>>(slotsWs, updWs, WiT, WhT, gru_bi, gru_bh,
                                  ln_mlp_g, ln_mlp_b, M1T, mlp_b1, M2T, mlp_b2, hidWs);
  }

  k_out<<<64, 256, 0, stream>>>(slotsWs, ln_out_g, ln_out_b, OwT, out_b, out);
}
